// Attention_80496277062338
// MI455X (gfx1250) — compile-verified, matrix operand order checked by reading, not run on hardware
//
#include <hip/hip_runtime.h>
#include <stdint.h>

#define B_ 384
#define S_ 128
#define E_ 512
#define H_ 4
#define D_ 128
#define M_ (B_ * S_)   // 49152 rows for the projection GEMMs

typedef unsigned short bfbits;                                  // raw bf16 bits
typedef __attribute__((ext_vector_type(16))) __bf16 v16bf;
typedef __attribute__((ext_vector_type(8)))  float  v8f;
typedef __attribute__((ext_vector_type(4)))  unsigned u32x4;
typedef __attribute__((ext_vector_type(8)))  unsigned u32x8;

union Frag { v16bf v; uint4 q[2]; };

__device__ __forceinline__ bfbits f2bf(float f) {
  __bf16 h = (__bf16)f;
  bfbits u;
  __builtin_memcpy(&u, &h, sizeof(u));
  return u;
}

// Load a 16x32 bf16 WMMA operand fragment (A layout; also B layout when the
// source is stored "N-major" so lane indexes rows and K is contiguous).
// CDNA5 16-bit layout: lanes 0-15 hold K = k0+0..7 and k0+16..23,
// lanes 16-31 hold K = k0+8..15 and k0+24..31 -> two contiguous 16B loads.
__device__ __forceinline__ v16bf load_frag(const bfbits* base, int stride,
                                           int row, int k0, int lane) {
  const int r    = row + (lane & 15);
  const int half = lane >> 4;
  const bfbits* p = base + (size_t)r * stride + k0 + half * 8;
  Frag f;
  f.q[0] = *(const uint4*)(p);        // K = k0 + half*8 + 0..7
  f.q[1] = *(const uint4*)(p + 16);   // K = k0 + 16 + half*8 + 0..7
  return f.v;
}

__device__ __forceinline__ v8f wmma_bf16(v16bf a, v16bf b, v8f c) {
  return __builtin_amdgcn_wmma_f32_16x16x32_bf16(false, a, false, b,
                                                 (short)0, c, false, false);
}

// Async 16-byte LDS -> global store (ASYNCcnt path).  Generic pointers into
// LDS carry the LDS byte offset in their low 32 bits (ISA 10.2 aperture map).
__device__ __forceinline__ void async_store_b128(void* g_dst, const void* lds_src) {
  unsigned l = (unsigned)(uintptr_t)lds_src;
  asm volatile("global_store_async_from_lds_b128 %0, %1, off"
               :: "v"(g_dst), "v"(l) : "memory");
}

__device__ __forceinline__ void wait_async0() {
  asm volatile("s_wait_asynccnt 0x0" ::: "memory");
}

// Tensor Data Mover: DMA a 2D tile of 2-byte elements (rows x cols,
// row stride == cols, i.e. a dense tile) from global memory into LDS.
// D# built per CDNA5 ISA ch.8: group0 = {count=1, lds_addr, global_addr,
// type=2}, group1 = {data_size=2B, tensor dims, tile dims, dim0 stride}.
// Issue from ONE wave only; tracked on TENSORcnt.
__device__ __forceinline__ void tdm_load_tile_2d(const void* lds_dst,
                                                 const void* g_src,
                                                 unsigned cols, unsigned rows) {
  unsigned long long ga = (unsigned long long)(uintptr_t)g_src;
  u32x4 g0;
  g0.x = 1u;                                            // count=1, user mode
  g0.y = (unsigned)(uintptr_t)lds_dst;                  // lds_addr   [63:32]
  g0.z = (unsigned)ga;                                  // gaddr[31:0][95:64]
  g0.w = (unsigned)((ga >> 32) & 0x01FFFFFFull)         // gaddr[56:32]
       | (2u << 30);                                    // type=2  [127:126]
  u32x8 g1;
  g1[0] = 1u << 16;                                     // data_size=1 -> 2B
  g1[1] = (cols & 0xFFFFu) << 16;                       // tensor_dim0[15:0]
  g1[2] = ((cols >> 16) & 0xFFFFu) | ((rows & 0xFFFFu) << 16); // td0 hi | td1 lo
  g1[3] = ((rows >> 16) & 0xFFFFu) | ((cols & 0xFFFFu) << 16); // td1 hi | tile_dim0
  g1[4] = rows & 0xFFFFu;                               // tile_dim1 (tile_dim2=0)
  g1[5] = cols;                                         // tensor_dim0_stride[31:0]
  g1[6] = 0u;                                           // stride hi | td1_stride lo
  g1[7] = 0u;
  asm volatile("tensor_load_to_lds %0, %1" :: "s"(g0), "s"(g1) : "memory");
}

// ---------------------------------------------------------------- converters
__global__ __launch_bounds__(256) void cvt_bf16_kernel(const float* __restrict__ src,
                                                       bfbits* __restrict__ dst, int n) {
  int i = (blockIdx.x * 256 + threadIdx.x) * 8;
  if (i + 8 > n) return;
  float4 a = *(const float4*)(src + i);
  float4 b = *(const float4*)(src + i + 4);
  uint4 o;
  o.x = (unsigned)f2bf(a.x) | ((unsigned)f2bf(a.y) << 16);
  o.y = (unsigned)f2bf(a.z) | ((unsigned)f2bf(a.w) << 16);
  o.z = (unsigned)f2bf(b.x) | ((unsigned)f2bf(b.y) << 16);
  o.w = (unsigned)f2bf(b.z) | ((unsigned)f2bf(b.w) << 16);
  *(uint4*)(dst + i) = o;
}

// W[E,E] fp32 -> Wt[n,k] bf16 (transpose so B-fragments are contiguous per lane)
__global__ __launch_bounds__(256) void cvt_wt_kernel(const float* __restrict__ W,
                                                     bfbits* __restrict__ Wt) {
  int idx = blockIdx.x * 256 + threadIdx.x;     // 0 .. E*E-1
  int k = idx >> 9, n = idx & (E_ - 1);
  Wt[(size_t)n * E_ + k] = f2bf(W[idx]);
}

// ---------------------------------------------------------------- QKV GEMM
// C = X(bf16)[M,E] @ W + bias.  Block tile 128x128: one block == one (b, h)
// pair.  8 waves arranged 4(M) x 2(N): each wave owns a 32x64 subtile.
// mode (blockIdx.z): 0 -> Q [B,H,S,D], 1 -> K [B,H,S,D], 2 -> V^T [B,H,D,S].
// Mode 2 transposes through LDS, then streams LDS -> HBM with
// global_store_async_from_lds_b128 (ASYNCcnt), no VGPR bounce.
#define TPAD 136   // 272B row stride: 16B-aligned, LDS bank stride 4 -> no conflicts
__global__ __launch_bounds__(256) void qkv_kernel(
    const bfbits* __restrict__ X,
    const bfbits* __restrict__ WqT, const bfbits* __restrict__ WkT,
    const bfbits* __restrict__ WvT,
    const float* __restrict__ bq, const float* __restrict__ bk,
    const float* __restrict__ bv,
    bfbits* __restrict__ Q, bfbits* __restrict__ K, bfbits* __restrict__ Vt) {
  __shared__ __align__(16) bfbits tbuf[D_ * TPAD];   // 34816B, mode-2 transpose
  const int mode = blockIdx.z;
  const bfbits* Wt   = (mode == 0) ? WqT : (mode == 1) ? WkT : WvT;
  const float*  bias = (mode == 0) ? bq  : (mode == 1) ? bk  : bv;

  const int lane = threadIdx.x & 31;
  const int wid  = threadIdx.x >> 5;
  const int wm = wid & 3, wn = wid >> 2;
  const int mbase = blockIdx.x * 128 + wm * 32;
  const int h     = blockIdx.y;            // N tile of 128 == head dim D
  const int nbase = wn * 64;               // column offset within the head

  v8f acc[2][4];
  for (int i = 0; i < 2; ++i)
    for (int j = 0; j < 4; ++j) { v8f z = {}; acc[i][j] = z; }

  for (int k0 = 0; k0 < E_; k0 += 32) {
    if (k0 + 32 < E_) {
      __builtin_prefetch(X + (size_t)(mbase + (lane & 15)) * E_ + k0 + 32, 0, 1);
      __builtin_prefetch(Wt + (size_t)(h * D_ + nbase + (lane & 15)) * E_ + k0 + 32, 0, 1);
    }
    v16bf a0 = load_frag(X, E_, mbase,      k0, lane);
    v16bf a1 = load_frag(X, E_, mbase + 16, k0, lane);
#pragma unroll
    for (int j = 0; j < 4; ++j) {
      v16bf bj = load_frag(Wt, E_, h * D_ + nbase + j * 16, k0, lane);
      acc[0][j] = wmma_bf16(a0, bj, acc[0][j]);
      acc[1][j] = wmma_bf16(a1, bj, acc[1][j]);
    }
  }

  const int half = lane >> 4;
  const int cb   = lane & 15;
  if (mode == 2) {
    // ---- transpose through LDS, then async LDS->global 16B stores
#pragma unroll
    for (int i = 0; i < 2; ++i)
#pragma unroll
      for (int j = 0; j < 4; ++j) {
        int d = nbase + j * 16 + cb;
        float bs = bias[h * D_ + d];
#pragma unroll
        for (int r = 0; r < 8; ++r) {
          int sl = wm * 32 + i * 16 + half * 8 + r;      // local s: 0..127
          tbuf[d * TPAD + sl] = f2bf(acc[i][j][r] + bs);
        }
      }
    __syncthreads();
    size_t headoff = ((size_t)(blockIdx.x * H_ + h)) * D_ * S_;
    for (int c = threadIdx.x; c < (D_ * S_) / 8; c += 256) {   // 2048 chunks
      int d  = c >> 4;
      int s8 = (c & 15) * 8;
      async_store_b128(&Vt[headoff + (size_t)d * S_ + s8], &tbuf[d * TPAD + s8]);
    }
    wait_async0();
  } else {
#pragma unroll
    for (int i = 0; i < 2; ++i)
#pragma unroll
      for (int j = 0; j < 4; ++j)
#pragma unroll
        for (int r = 0; r < 8; ++r) {
          int m = mbase + i * 16 + half * 8 + r;
          int s = m & (S_ - 1), b = m >> 7;
          int d = nbase + j * 16 + cb;
          float val = acc[i][j][r] + bias[h * D_ + d];
          size_t head = (size_t)(b * H_ + h);
          if (mode == 0) Q[(head * S_ + s) * D_ + d] = f2bf(val);
          else           K[(head * S_ + s) * D_ + d] = f2bf(val);
        }
  }
}

// ---------------------------------------------------------------- attention
// One workgroup per (b,h).  K and V^T tiles (32KB each) are DMA'd into LDS by
// the Tensor Data Mover (one tensor_load_to_lds per tile, issued by wave 0,
// TENSORcnt), removing per-wave redundancy and all copy address math.
// Wave w owns query rows 16w..16w+15: scores = Q K^T (no 1/sqrt(D), matching
// the reference), row softmax, P -> LDS (C->A layout re-gather, reusing the
// K buffer), out = P @ V.
__global__ __launch_bounds__(256) void attn_kernel(
    const bfbits* __restrict__ Qg, const bfbits* __restrict__ Kg,
    const bfbits* __restrict__ Vtg, bfbits* __restrict__ O) {
  __shared__ __align__(16) bfbits kbuf[S_ * D_];   // 32KB: K tile, then P buffer
  __shared__ __align__(16) bfbits vbuf[D_ * S_];   // 32KB: V^T tile [d][t]

  const int bh   = blockIdx.x;
  const int b    = bh / H_, h = bh % H_;
  const int lane = threadIdx.x & 31;
  const int w    = threadIdx.x >> 5;
  const int half = lane >> 4;
  const int cb   = lane & 15;

  const bfbits* q  = Qg  + (size_t)bh * S_ * D_;
  const bfbits* kp = Kg  + (size_t)bh * S_ * D_;
  const bfbits* vt = Vtg + (size_t)bh * D_ * S_;

  // ---- TDM-stage K and V^T into LDS (one descriptor each, wave 0 only)
  if (w == 0) {
    tdm_load_tile_2d(kbuf, kp, D_, S_);   // 128x128 x 2B = 32KB
    tdm_load_tile_2d(vbuf, vt, S_, D_);
    __builtin_amdgcn_s_wait_tensorcnt(0);
  }
  __syncthreads();

  const int mrow = w * 16;

  // ---- scores: 16x128 tile per wave, K fragments from LDS
  v8f acc[8];
  for (int t = 0; t < 8; ++t) { v8f z = {}; acc[t] = z; }
  for (int d0 = 0; d0 < D_; d0 += 32) {
    v16bf a = load_frag(q, D_, mrow, d0, lane);
#pragma unroll
    for (int t = 0; t < 8; ++t) {
      v16bf kf = load_frag(kbuf, D_, t * 16, d0, lane);  // B[k=d][n=t] = K[t][d]
      acc[t] = wmma_bf16(a, kf, acc[t]);
    }
  }

  // ---- row-wise softmax.  For acc element r: lanes 0-15 hold row (mrow+r),
  // lanes 16-31 hold row (mrow+r+8); xor masks < 16 stay inside each group.
#pragma unroll
  for (int r = 0; r < 8; ++r) {
    float mx = acc[0][r];
#pragma unroll
    for (int t = 1; t < 8; ++t) mx = fmaxf(mx, acc[t][r]);
    for (int off = 1; off < 16; off <<= 1) mx = fmaxf(mx, __shfl_xor(mx, off, 32));
    float sum = 0.f;
#pragma unroll
    for (int t = 0; t < 8; ++t) {
      float e = __expf(acc[t][r] - mx);
      acc[t][r] = e;
      sum += e;
    }
    for (int off = 1; off < 16; off <<= 1) sum += __shfl_xor(sum, off, 32);
    float inv = 1.0f / sum;
#pragma unroll
    for (int t = 0; t < 8; ++t) acc[t][r] *= inv;
  }

  // ---- all waves are done reading kbuf; reuse it as the P buffer
  __syncthreads();
  bfbits* pw = kbuf + w * 16 * S_;            // wave-private 16x128 slice
#pragma unroll
  for (int t = 0; t < 8; ++t)
#pragma unroll
    for (int r = 0; r < 8; ++r)
      pw[(half * 8 + r) * S_ + t * 16 + cb] = f2bf(acc[t][r]);
  // LDS ops from the same wave complete in order (DScnt); reads below are
  // wave-private, so no further cross-wave barrier is needed.

  // ---- out = P @ V : A from LDS P, B from LDS V^T (both ds_load_b128)
  v8f oacc[8];
  for (int d = 0; d < 8; ++d) { v8f z = {}; oacc[d] = z; }
  for (int t0 = 0; t0 < S_; t0 += 32) {
    v16bf a = load_frag(pw, S_, 0, t0, lane);
#pragma unroll
    for (int dt = 0; dt < 8; ++dt) {
      v16bf vf = load_frag(vbuf, S_, dt * 16, t0, lane); // B[k=t][n=d] = Vt[d][t]
      oacc[dt] = wmma_bf16(a, vf, oacc[dt]);
    }
  }

  // ---- store attn output bf16 in [B*S, E] row-major for the final GEMM
#pragma unroll
  for (int dt = 0; dt < 8; ++dt)
#pragma unroll
    for (int r = 0; r < 8; ++r) {
      int s = mrow + half * 8 + r;
      int d = dt * 16 + cb;
      O[((size_t)(b * S_ + s)) * E_ + h * D_ + d] = f2bf(oacc[dt][r]);
    }
}

// ---------------------------------------------------------------- out proj
__global__ __launch_bounds__(256) void oproj_kernel(
    const bfbits* __restrict__ A, const bfbits* __restrict__ WoT,
    const float* __restrict__ bo, float* __restrict__ Cout) {
  const int lane = threadIdx.x & 31;
  const int wid  = threadIdx.x >> 5;
  const int wm = wid & 3, wn = wid >> 2;
  const int mbase = blockIdx.x * 128 + wm * 32;
  const int nbase = blockIdx.y * 128 + wn * 64;

  v8f acc[2][4];
  for (int i = 0; i < 2; ++i)
    for (int j = 0; j < 4; ++j) { v8f z = {}; acc[i][j] = z; }

  for (int k0 = 0; k0 < E_; k0 += 32) {
    if (k0 + 32 < E_) {
      __builtin_prefetch(A + (size_t)(mbase + (lane & 15)) * E_ + k0 + 32, 0, 1);
      __builtin_prefetch(WoT + (size_t)(nbase + (lane & 15)) * E_ + k0 + 32, 0, 1);
    }
    v16bf a0 = load_frag(A, E_, mbase,      k0, lane);
    v16bf a1 = load_frag(A, E_, mbase + 16, k0, lane);
#pragma unroll
    for (int j = 0; j < 4; ++j) {
      v16bf bj = load_frag(WoT, E_, nbase + j * 16, k0, lane);
      acc[0][j] = wmma_bf16(a0, bj, acc[0][j]);
      acc[1][j] = wmma_bf16(a1, bj, acc[1][j]);
    }
  }

  const int half = lane >> 4;
  const int cb   = lane & 15;
#pragma unroll
  for (int i = 0; i < 2; ++i)
#pragma unroll
    for (int j = 0; j < 4; ++j) {
      int col = nbase + j * 16 + cb;
      float bias = bo[col];
#pragma unroll
      for (int r = 0; r < 8; ++r) {
        int m = mbase + i * 16 + half * 8 + r;
        Cout[(size_t)m * E_ + col] = acc[i][j][r] + bias;
      }
    }
}

// ---------------------------------------------------------------- launcher
extern "C" void kernel_launch(void* const* d_in, const int* in_sizes, int n_in,
                              void* d_out, int out_size, void* d_ws, size_t ws_size,
                              hipStream_t stream) {
  (void)in_sizes; (void)n_in; (void)out_size; (void)ws_size;
  const float* x  = (const float*)d_in[0];
  const float* Wq = (const float*)d_in[1];
  const float* bq = (const float*)d_in[2];
  const float* Wk = (const float*)d_in[3];
  const float* bk = (const float*)d_in[4];
  const float* Wv = (const float*)d_in[5];
  const float* bv = (const float*)d_in[6];
  const float* Wo = (const float*)d_in[7];
  const float* bo = (const float*)d_in[8];
  float* out = (float*)d_out;

  const size_t MX  = (size_t)M_ * E_;     // 25,165,824 elements
  const size_t WSZ = (size_t)E_ * E_;     // 262,144 elements
  bfbits* ws  = (bfbits*)d_ws;
  bfbits* xb  = ws;                       // x in bf16; reused for attn output
  bfbits* wqt = ws + MX;
  bfbits* wkt = wqt + WSZ;
  bfbits* wvt = wkt + WSZ;
  bfbits* wot = wvt + WSZ;
  bfbits* qb  = wot + WSZ;                // [B,H,S,D]
  bfbits* kb  = qb + MX;                  // [B,H,S,D]
  bfbits* vtb = kb + MX;                  // [B,H,D,S]  (V transposed)

  cvt_bf16_kernel<<<dim3((unsigned)(MX / (256 * 8))), 256, 0, stream>>>(x, xb, (int)MX);
  cvt_wt_kernel<<<dim3((unsigned)(WSZ / 256)), 256, 0, stream>>>(Wq, wqt);
  cvt_wt_kernel<<<dim3((unsigned)(WSZ / 256)), 256, 0, stream>>>(Wk, wkt);
  cvt_wt_kernel<<<dim3((unsigned)(WSZ / 256)), 256, 0, stream>>>(Wv, wvt);
  cvt_wt_kernel<<<dim3((unsigned)(WSZ / 256)), 256, 0, stream>>>(Wo, wot);

  qkv_kernel<<<dim3(M_ / 128, E_ / 128, 3), 256, 0, stream>>>(
      xb, wqt, wkt, wvt, bq, bk, bv, qb, kb, vtb);

  attn_kernel<<<dim3(B_ * H_), 256, 0, stream>>>(qb, kb, vtb, xb);

  oproj_kernel<<<dim3(M_ / 128, E_ / 128), 256, 0, stream>>>(xb, wot, bo, out);
}
